// MINKUNetBackbone_28295244546744
// MI455X (gfx1250) — matrix operand order repaired and checked by reading, not compiled, not run on hardware
//
#include <hip/hip_runtime.h>

// ---------------------------------------------------------------------------
// CDNA5 (gfx1250) MinkUNet backbone:
//  - all contractions via v_wmma_f32_16x16x32_bf16 (fp32 accumulate)
//  - weight tiles staged cooperatively into LDS (async global->LDS engine,
//    ASYNCcnt-tracked, double-buffered across the 27-tap loop)
//  - 8 waves/block share one n-chunk so B tiles are read from LDS 8x
// ---------------------------------------------------------------------------

typedef __attribute__((ext_vector_type(16))) __bf16 v16bf;
typedef __attribute__((ext_vector_type(8)))  __bf16 v8bf;
typedef __attribute__((ext_vector_type(8)))  float  v8f;
typedef __attribute__((ext_vector_type(4)))  int    v4i;

union ABPack { v16bf v; v8bf h[2]; };

#if defined(__has_builtin)
#  if __has_builtin(__builtin_amdgcn_global_load_async_to_lds_b128)
#    define HAVE_ASYNC_LDS 1
#  endif
#endif
#ifndef HAVE_ASYNC_LDS
#  define HAVE_ASYNC_LDS 0
#endif

// 16-byte global -> LDS copy (async engine if available, else through VGPRs)
__device__ __forceinline__ void copy16_to_lds(const char* gsrc, char* ldst) {
#if HAVE_ASYNC_LDS
  typedef __attribute__((address_space(1))) v4i* gptr_t;   // global v4i*
  typedef __attribute__((address_space(3))) v4i* lptr_t;   // LDS v4i*
  __builtin_amdgcn_global_load_async_to_lds_b128(
      (gptr_t)(char*)gsrc, (lptr_t)ldst, 0, 0);
#else
  uint4 v = *(const uint4*)gsrc;
  *(uint4*)ldst = v;   // ds_store_b128 (addrspace inferred)
#endif
}

__device__ __forceinline__ void wait_async0() {
#if HAVE_ASYNC_LDS
#  if __has_builtin(__builtin_amdgcn_s_wait_asynccnt)
  __builtin_amdgcn_s_wait_asynccnt(0);
#  else
  asm volatile("s_wait_asynccnt 0x0" ::: "memory");
#  endif
#endif
}

// A-tile (16x32 bf16) per-lane load (contiguous k order, matches packed B):
//   lanes 0-15 : M=lane,    K = {0..15}
//   lanes 16-31: M=lane-16, K = {16..31}
__device__ __forceinline__ v16bf load_a_tile(const __bf16* __restrict__ row, int cbase, int hi) {
  ABPack a;
  a.h[0] = *(const v8bf*)(row + cbase + hi * 16);
  a.h[1] = *(const v8bf*)(row + cbase + 8 + hi * 16);
  return a.v;
}

__device__ __forceinline__ v8f wmma_bf16(v16bf a, v16bf b, v8f c) {
  return __builtin_amdgcn_wmma_f32_16x16x32_bf16(false, a, false, b, (short)0, c, false, false);
}

// ---------------------------------------------------------------------------
// Preprocessing kernels
// ---------------------------------------------------------------------------

// Pad feats [N,3] f32 -> [N,32] bf16 (channels 3..31 zero)
__global__ void pad_feats_kernel(const float* __restrict__ f, __bf16* __restrict__ o, int N) {
  int i = blockIdx.x * blockDim.x + threadIdx.x;
  if (i >= N * 32) return;
  int n = i >> 5, c = i & 31;
  float v = (c < 3) ? f[(size_t)n * 3 + c] : 0.f;
  o[i] = (__bf16)v;
}

// Pack W[K,CinReal,Cout] f32 -> bf16 tiles in WMMA B-operand lane order.
// Tile id = (k*cinBlk + cblk)*NT + nblk; within tile: lane (0..31) x slot j (0..15):
//   cin  = cblk*32 + (lane>>4)*16 + j,  cout = nblk*16 + (lane&15)
__global__ void pack_w_kernel(const float* __restrict__ W, __bf16* __restrict__ o,
                              int K, int cinReal, int cinBlk, int NT, int Cout) {
  int i = blockIdx.x * blockDim.x + threadIdx.x;
  int total = K * cinBlk * NT * 512;
  if (i >= total) return;
  int tile = i >> 9;
  int e    = i & 511;
  int lane = e >> 4, j = e & 15;
  int nblk = tile % NT;
  int t2   = tile / NT;
  int cblk = t2 % cinBlk;
  int k    = t2 / cinBlk;
  int cin  = cblk * 32 + (lane >> 4) * 16 + j;
  int cout = nblk * 16 + (lane & 15);
  float v = (cin < cinReal) ? W[((size_t)k * cinReal + cin) * Cout + cout] : 0.f;
  o[i] = (__bf16)v;
}

// ---------------------------------------------------------------------------
// Sparse conv: out[m,:] = (res? res[m,:]:0) + relu( sum_k sum_c feat[nbr[m,k],c] * W[k,c,:] )
// Block = 8 m-tiles x one n-chunk (<=64 couts). Weight tiles for each k staged
// into LDS (double-buffered, async), shared by all 8 waves.
// ---------------------------------------------------------------------------
__global__ __launch_bounds__(256) void conv_wmma_kernel(
    const __bf16* __restrict__ feat, const int* __restrict__ nbr,
    const __bf16* __restrict__ Wp, const __bf16* __restrict__ res,
    __bf16* __restrict__ out, int M, int K, int cinBlk, int ldf, int Cout)
{
  extern __shared__ __align__(32) char smem[];
  const int NT      = Cout >> 4;
  const int nchunks = (NT + 3) >> 2;
  const int mtiles  = (M + 15) >> 4;
  const int mgroup  = (int)blockIdx.x / nchunks;
  const int n0      = ((int)blockIdx.x % nchunks) * 4;
  const int nt      = (NT - n0 < 4) ? (NT - n0) : 4;
  const int wid  = (int)(threadIdx.x >> 5);
  const int lane = (int)(threadIdx.x & 31);

  int mtile = mgroup * 8 + wid;
  const bool valid = (mtile < mtiles);
  if (!valid) mtile = mtiles - 1;          // clamp: waves must stay for barriers
  const int hi = lane >> 4;
  int m = mtile * 16 + (lane & 15);
  if (m >= M) m = M - 1;                   // clamp: uniform control flow for WMMA

  const int bufStride = cinBlk * 4096;     // bytes per LDS buffer (max nt = 4)
  const int segBytes  = nt * 1024;         // bytes staged per cin-block

  // stage weight tiles for kernel-tap k into LDS buffer `buf`
  auto stage = [&](int k, int buf) {
    char* dst = smem + buf * bufStride;
    for (int cb = 0; cb < cinBlk; ++cb) {
      const char* src = (const char*)Wp + ((size_t)(k * cinBlk + cb) * NT + n0) * 1024;
      char* d = dst + cb * segBytes;
      for (int ofs = (int)threadIdx.x * 16; ofs < segBytes; ofs += 256 * 16)
        copy16_to_lds(src + ofs, d + ofs);
    }
  };

  v8f a0 = {}, a1 = {}, a2 = {}, a3 = {};

  stage(0, 0);
  for (int k = 0; k < K; ++k) {
    const int cur = k & 1;
    wait_async0();                          // my async copies for buffer `cur` done
    __syncthreads();                        // all waves' copies done
    if (k + 1 < K) stage(k + 1, cur ^ 1);   // prefetch next tap while computing

    const int gi = nbr[(size_t)m * K + k];
    const __bf16* arow = feat + (size_t)gi * ldf;
    const char* bbuf = smem + cur * bufStride + (size_t)lane * 32;
    for (int cb = 0; cb < cinBlk; ++cb) {
      v16bf av = load_a_tile(arow, cb * 32, hi);
      const char* tb = bbuf + cb * segBytes;
      v16bf bv0 = *(const v16bf*)tb;
      a0 = wmma_bf16(av, bv0, a0);
      if (nt > 1) { v16bf bv = *(const v16bf*)(tb + 1024); a1 = wmma_bf16(av, bv, a1); }
      if (nt > 2) { v16bf bv = *(const v16bf*)(tb + 2048); a2 = wmma_bf16(av, bv, a2); }
      if (nt > 3) { v16bf bv = *(const v16bf*)(tb + 3072); a3 = wmma_bf16(av, bv, a3); }
    }
    __syncthreads();                        // done reading `cur` before it is refilled
  }

  const int ncol = lane & 15;
#pragma unroll
  for (int r = 0; r < 8; ++r) {
    const int mrow = mtile * 16 + r + hi * 8;   // C/D layout: VGPR r -> row r (+8 hi lanes)
    if (!valid || mrow >= M) continue;
    const size_t ro = (size_t)mrow * Cout;
    { float v = a0[r]; v = v > 0.f ? v : 0.f; int n = (n0 + 0) * 16 + ncol;
      if (res) v += (float)res[ro + n]; out[ro + n] = (__bf16)v; }
    if (nt > 1) { float v = a1[r]; v = v > 0.f ? v : 0.f; int n = (n0 + 1) * 16 + ncol;
      if (res) v += (float)res[ro + n]; out[ro + n] = (__bf16)v; }
    if (nt > 2) { float v = a2[r]; v = v > 0.f ? v : 0.f; int n = (n0 + 2) * 16 + ncol;
      if (res) v += (float)res[ro + n]; out[ro + n] = (__bf16)v; }
    if (nt > 3) { float v = a3[r]; v = v > 0.f ? v : 0.f; int n = (n0 + 3) * 16 + ncol;
      if (res) v += (float)res[ro + n]; out[ro + n] = (__bf16)v; }
  }
}

// ---------------------------------------------------------------------------
// Concat linear: out[m,:] = act( [srcA[gidx[m]], srcB[m]] @ W )
// All weight tiles for the block's n-chunk staged into LDS once (async).
// If outf != null: fp32 transposed store out[(m/S)*Cout + n, m%S], no relu.
// ---------------------------------------------------------------------------
__global__ __launch_bounds__(256) void lin_wmma_kernel(
    const __bf16* __restrict__ srcA, const int* __restrict__ gidx, int lda, int caBlk,
    const __bf16* __restrict__ srcB, int ldb, int cbBlk,
    const __bf16* __restrict__ Wp,
    __bf16* __restrict__ outb, float* __restrict__ outf,
    int M, int Cout, int S)
{
  extern __shared__ __align__(32) char smem[];
  const int NT      = Cout >> 4;
  const int nchunks = (NT + 3) >> 2;
  const int mtiles  = (M + 15) >> 4;
  const int mgroup  = (int)blockIdx.x / nchunks;
  const int n0      = ((int)blockIdx.x % nchunks) * 4;
  const int nt      = (NT - n0 < 4) ? (NT - n0) : 4;
  const int wid  = (int)(threadIdx.x >> 5);
  const int lane = (int)(threadIdx.x & 31);

  int mtile = mgroup * 8 + wid;
  const bool valid = (mtile < mtiles);
  if (!valid) mtile = mtiles - 1;
  const int hi = lane >> 4;
  int m = mtile * 16 + (lane & 15);
  if (m >= M) m = M - 1;

  const int totBlk   = caBlk + cbBlk;
  const int segBytes = nt * 1024;

  // one-shot stage of all weight tiles for this n-chunk
  for (int cb = 0; cb < totBlk; ++cb) {
    const char* src = (const char*)Wp + ((size_t)cb * NT + n0) * 1024;
    char* d = smem + cb * segBytes;
    for (int ofs = (int)threadIdx.x * 16; ofs < segBytes; ofs += 256 * 16)
      copy16_to_lds(src + ofs, d + ofs);
  }
  wait_async0();
  __syncthreads();

  const int ra = gidx ? gidx[m] : m;
  const __bf16* arow = srcA + (size_t)ra * lda;
  const __bf16* brow = srcB ? (srcB + (size_t)m * ldb) : (const __bf16*)0;

  v8f a0 = {}, a1 = {}, a2 = {}, a3 = {};
  const char* bbase = smem + (size_t)lane * 32;

  for (int cb = 0; cb < totBlk; ++cb) {
    const __bf16* rowp;
    int coff;
    if (cb < caBlk) { rowp = arow; coff = cb * 32; }
    else            { rowp = brow; coff = (cb - caBlk) * 32; }
    v16bf av = load_a_tile(rowp, coff, hi);
    const char* tb = bbase + cb * segBytes;
    v16bf bv0 = *(const v16bf*)tb;
    a0 = wmma_bf16(av, bv0, a0);
    if (nt > 1) { v16bf bv = *(const v16bf*)(tb + 1024); a1 = wmma_bf16(av, bv, a1); }
    if (nt > 2) { v16bf bv = *(const v16bf*)(tb + 2048); a2 = wmma_bf16(av, bv, a2); }
    if (nt > 3) { v16bf bv = *(const v16bf*)(tb + 3072); a3 = wmma_bf16(av, bv, a3); }
  }

  const int ncol = lane & 15;
#pragma unroll
  for (int r = 0; r < 8; ++r) {
    const int mrow = mtile * 16 + r + hi * 8;
    if (!valid || mrow >= M) continue;
    if (outf) {
      const int b = mrow / S, s = mrow - b * S;
      { float v = a0[r]; int n = (n0 + 0) * 16 + ncol; outf[((size_t)b * Cout + n) * S + s] = v; }
      if (nt > 1) { float v = a1[r]; int n = (n0 + 1) * 16 + ncol; outf[((size_t)b * Cout + n) * S + s] = v; }
      if (nt > 2) { float v = a2[r]; int n = (n0 + 2) * 16 + ncol; outf[((size_t)b * Cout + n) * S + s] = v; }
      if (nt > 3) { float v = a3[r]; int n = (n0 + 3) * 16 + ncol; outf[((size_t)b * Cout + n) * S + s] = v; }
    } else {
      const size_t ro = (size_t)mrow * Cout;
      { float v = a0[r]; v = v > 0.f ? v : 0.f; int n = (n0 + 0) * 16 + ncol; outb[ro + n] = (__bf16)v; }
      if (nt > 1) { float v = a1[r]; v = v > 0.f ? v : 0.f; int n = (n0 + 1) * 16 + ncol; outb[ro + n] = (__bf16)v; }
      if (nt > 2) { float v = a2[r]; v = v > 0.f ? v : 0.f; int n = (n0 + 2) * 16 + ncol; outb[ro + n] = (__bf16)v; }
      if (nt > 3) { float v = a3[r]; v = v > 0.f ? v : 0.f; int n = (n0 + 3) * 16 + ncol; outb[ro + n] = (__bf16)v; }
    }
  }
}

// ---------------------------------------------------------------------------
// Launcher
// ---------------------------------------------------------------------------
extern "C" void kernel_launch(void* const* d_in, const int* in_sizes, int n_in,
                              void* d_out, int out_size, void* d_ws, size_t ws_size,
                              hipStream_t stream) {
  const int N0 = 100000, N1 = 50000, N2 = 25000, N3 = 12500;
  const int K = 27, KD = 8, S = 1024, COUT = 512, SEEDS = 8 * 1024;

  const float* feats   = (const float*)d_in[0];
  const int*   nbr0    = (const int*)d_in[1];
  const int*   nbr1    = (const int*)d_in[2];
  const int*   nbr2    = (const int*)d_in[3];
  const int*   nbr3    = (const int*)d_in[4];
  const int*   down1   = (const int*)d_in[5];
  const int*   down2   = (const int*)d_in[6];
  const int*   down3   = (const int*)d_in[7];
  const int*   up2     = (const int*)d_in[8];
  const int*   up1     = (const int*)d_in[9];
  const int*   up0     = (const int*)d_in[10];
  const int*   seedIdx = (const int*)d_in[11];
  const float* W_stem = (const float*)d_in[12];
  const float* W_b0   = (const float*)d_in[13];
  const float* W_d1   = (const float*)d_in[14];
  const float* W_b1   = (const float*)d_in[15];
  const float* W_d2   = (const float*)d_in[16];
  const float* W_b2   = (const float*)d_in[17];
  const float* W_d3   = (const float*)d_in[18];
  const float* W_b3   = (const float*)d_in[19];
  const float* W_u2   = (const float*)d_in[20];
  const float* W_u1   = (const float*)d_in[21];
  const float* W_u0   = (const float*)d_in[22];
  const float* W_out  = (const float*)d_in[23];

  // ---- workspace carve-up (256B aligned) ----
  char* ws = (char*)d_ws;
  size_t off = 0;
  auto alloc = [&](size_t bytes) -> char* {
    char* p = ws + off;
    off += (bytes + 255) & ~(size_t)255;
    return p;
  };
  __bf16* f0p = (__bf16*)alloc((size_t)N0 * 32 * 2);
  __bf16* x0  = (__bf16*)alloc((size_t)N0 * 32 * 2);
  __bf16* x0r = (__bf16*)alloc((size_t)N0 * 32 * 2);
  __bf16* x1  = (__bf16*)alloc((size_t)N1 * 64 * 2);
  __bf16* x1r = (__bf16*)alloc((size_t)N1 * 64 * 2);
  __bf16* x2  = (__bf16*)alloc((size_t)N2 * 128 * 2);
  __bf16* x2r = (__bf16*)alloc((size_t)N2 * 128 * 2);
  __bf16* x3  = (__bf16*)alloc((size_t)N3 * 256 * 2);
  __bf16* x3r = (__bf16*)alloc((size_t)N3 * 256 * 2);
  __bf16* u2  = (__bf16*)alloc((size_t)N2 * 256 * 2);
  __bf16* u1  = (__bf16*)alloc((size_t)N1 * 128 * 2);
  __bf16* u0  = (__bf16*)alloc((size_t)N0 * 96 * 2);
  auto wtiles = [&](int Kk, int cinBlk, int NT) -> __bf16* {
    return (__bf16*)alloc((size_t)Kk * cinBlk * NT * 512 * 2);
  };
  __bf16* wpStem = wtiles(K, 1, 2);
  __bf16* wpB0   = wtiles(K, 1, 2);
  __bf16* wpD1   = wtiles(KD, 1, 4);
  __bf16* wpB1   = wtiles(K, 2, 4);
  __bf16* wpD2   = wtiles(KD, 2, 8);
  __bf16* wpB2   = wtiles(K, 4, 8);
  __bf16* wpD3   = wtiles(KD, 4, 16);
  __bf16* wpB3   = wtiles(K, 8, 16);
  __bf16* wpU2   = wtiles(1, 12, 16);
  __bf16* wpU1   = wtiles(1, 10, 8);
  __bf16* wpU0   = wtiles(1, 5, 6);
  __bf16* wpOut  = wtiles(1, 3, 32);
  (void)ws_size; (void)n_in; (void)in_sizes; (void)out_size;

  // ---- preprocessing: pad feats + pack all weights to WMMA B-tile layout ----
  {
    int tot = N0 * 32;
    pad_feats_kernel<<<(tot + 255) / 256, 256, 0, stream>>>(feats, f0p, N0);
  }
  auto pack = [&](const float* W, __bf16* o, int Kk, int cinReal, int cinBlk, int NT, int Cout) {
    int tot = Kk * cinBlk * NT * 512;
    pack_w_kernel<<<(tot + 255) / 256, 256, 0, stream>>>(W, o, Kk, cinReal, cinBlk, NT, Cout);
  };
  pack(W_stem, wpStem, K, 3,   1,  2,  32);
  pack(W_b0,   wpB0,   K, 32,  1,  2,  32);
  pack(W_d1,   wpD1,   KD, 32, 1,  4,  64);
  pack(W_b1,   wpB1,   K, 64,  2,  4,  64);
  pack(W_d2,   wpD2,   KD, 64, 2,  8,  128);
  pack(W_b2,   wpB2,   K, 128, 4,  8,  128);
  pack(W_d3,   wpD3,   KD, 128, 4, 16, 256);
  pack(W_b3,   wpB3,   K, 256, 8, 16, 256);
  pack(W_u2,   wpU2,   1, 384, 12, 16, 256);
  pack(W_u1,   wpU1,   1, 320, 10, 8,  128);
  pack(W_u0,   wpU0,   1, 160, 5,  6,  96);
  pack(W_out,  wpOut,  1, 96,  3,  32, 512);

  // ---- encoder (gather-conv WMMA, LDS-staged weights) ----
  auto conv = [&](const __bf16* feat, const int* nbr, const __bf16* Wp, const __bf16* res,
                  __bf16* out, int M, int Kk, int cinBlk, int ldf, int Cout) {
    int NT = Cout >> 4;
    int nchunks = (NT + 3) >> 2;
    int mgroups = (((M + 15) >> 4) + 7) >> 3;
    int shm = 2 * cinBlk * 4096;
    conv_wmma_kernel<<<mgroups * nchunks, 256, shm, stream>>>(feat, nbr, Wp, res, out,
                                                              M, Kk, cinBlk, ldf, Cout);
  };
  conv(f0p, nbr0,  wpStem, nullptr, x0,  N0, K,  1, 32,  32);   // stem (Cin padded 3->32)
  conv(x0,  nbr0,  wpB0,   x0,      x0r, N0, K,  1, 32,  32);   // x0 += relu(conv(x0))
  conv(x0r, down1, wpD1,   nullptr, x1,  N1, KD, 1, 32,  64);   // down 1
  conv(x1,  nbr1,  wpB1,   x1,      x1r, N1, K,  2, 64,  64);
  conv(x1r, down2, wpD2,   nullptr, x2,  N2, KD, 2, 64,  128);  // down 2
  conv(x2,  nbr2,  wpB2,   x2,      x2r, N2, K,  4, 128, 128);
  conv(x2r, down3, wpD3,   nullptr, x3,  N3, KD, 4, 128, 256);  // down 3
  conv(x3,  nbr3,  wpB3,   x3,      x3r, N3, K,  8, 256, 256);

  // ---- decoder (gather + concat + 1x1 WMMA, LDS-staged weights) ----
  auto lin = [&](const __bf16* a, const int* gidx, int lda, int caBlk,
                 const __bf16* b, int ldb, int cbBlk, const __bf16* Wp,
                 __bf16* outb, float* outf, int M, int Cout, int Sdim) {
    int NT = Cout >> 4;
    int nchunks = (NT + 3) >> 2;
    int mgroups = (((M + 15) >> 4) + 7) >> 3;
    int shm = (caBlk + cbBlk) * 4096;
    lin_wmma_kernel<<<mgroups * nchunks, 256, shm, stream>>>(a, gidx, lda, caBlk, b, ldb, cbBlk,
                                                             Wp, outb, outf, M, Cout, Sdim);
  };
  lin(x3r, up2, 256, 8, x2r, 128, 4, wpU2, u2, nullptr, N2, 256, 1);   // u2
  lin(u2,  up1, 256, 8, x1r, 64,  2, wpU1, u1, nullptr, N1, 128, 1);   // u1
  lin(u1,  up0, 128, 4, x0r, 32,  1, wpU0, u0, nullptr, N0, 96,  1);   // u0
  // final: only the 8192 seed rows, fp32 output, transposed to [B, 512, S]
  lin(u0, seedIdx, 96, 3, nullptr, 0, 0, wpOut, nullptr, (float*)d_out,
      SEEDS, COUT, S);
}
